// TransConvLayer_36653250904189
// MI455X (gfx1250) — matrix operand order repaired and checked by reading, not compiled, not run on hardware
//
#include <hip/hip_runtime.h>
#include <hip/hip_bf16.h>

typedef __bf16 bf16;
typedef __attribute__((ext_vector_type(16))) bf16  v16bf;
typedef __attribute__((ext_vector_type(8)))  float v8f;
typedef __attribute__((ext_vector_type(4)))  unsigned int v4u;
typedef __attribute__((ext_vector_type(8)))  int v8i;
typedef __attribute__((ext_vector_type(4)))  int v4i;

#define NROWS 32768
#define C 128
#define H 8
#define SPLITS 32
#define ROWS_PER_BLOCK (NROWS / SPLITS)   // 1024
#define TM 128
#define TILES (ROWS_PER_BLOCK / TM)       // 8
#define EPS_PHI 1e-6f
#define EPS_FP  1e-8f

#if __has_builtin(__builtin_amdgcn_tensor_load_to_lds)
#define HAS_TDM 1
#else
#define HAS_TDM 0
#endif

__device__ __forceinline__ v8f wmma_bf16(v16bf a, v16bf b, v8f c) {
  return __builtin_amdgcn_wmma_f32_16x16x32_bf16(false, a, false, b, (short)0, c, false, false);
}

// ---------------- fragment loaders (per ISA 16-bit WMMA layouts) ----------------
// A fragment from row-major bf16 LDS, leading dim 128 (contiguous -> ds_load_b128 x2)
__device__ __forceinline__ v16bf load_a_lds(const bf16* s, int m0, int k0, int lane) {
  int m = m0 + (lane & 15);
  int kb = k0 + 8 * (lane >> 4);
  const bf16* p = s + m * C + kb;
  v16bf a;
#pragma unroll
  for (int e = 0; e < 8; ++e) { a[e] = p[e]; a[e + 8] = p[16 + e]; }
  return a;
}
// A fragment directly from global f32 (row-major, ld=C), cvt to bf16
__device__ __forceinline__ v16bf load_a_gf32(const float* base, int m0, int k0, int lane) {
  int m = m0 + (lane & 15);
  int kb = k0 + 8 * (lane >> 4);
  const float* p = base + (size_t)m * C + kb;
  v16bf a;
#pragma unroll
  for (int e = 0; e < 8; ++e) { a[e] = (bf16)p[e]; a[e + 8] = (bf16)p[16 + e]; }
  return a;
}
// A fragment directly from global bf16 (row-major, ld=C)
__device__ __forceinline__ v16bf load_a_gbf(const bf16* base, int m0, int k0, int lane) {
  int m = m0 + (lane & 15);
  int kb = k0 + 8 * (lane >> 4);
  const bf16* p = base + (size_t)m * C + kb;
  v16bf a;
#pragma unroll
  for (int e = 0; e < 8; ++e) { a[e] = p[e]; a[e + 8] = p[16 + e]; }
  return a;
}
// B fragment where source stores B^T: B[k][n] = src[n*ldC + k] (contiguous in k)
__device__ __forceinline__ v16bf load_b_tr(const bf16* s, int k0, int n0, int lane) {
  int n = n0 + (lane & 15);
  int kb = k0 + 8 * (lane >> 4);
  const bf16* p = s + n * C + kb;
  v16bf b;
#pragma unroll
  for (int e = 0; e < 8; ++e) { b[e] = p[e]; b[e + 8] = p[16 + e]; }
  return b;
}

// packed 8x bf16 store (16B) — rows consecutive per lane
__device__ __forceinline__ void store8_bf16(bf16* dst, const float* v) {
  union { bf16 h[8]; uint4 u; } p;
#pragma unroll
  for (int r = 0; r < 8; ++r) p.h[r] = (bf16)v[r];
  *(uint4*)dst = p.u;
}

// ---------------- TDM helper: 1-D LDS <- global copy of n bf16 elements ----------
__device__ __forceinline__ unsigned lds_base_off() {
#if __has_builtin(__builtin_amdgcn_groupstaticsize)
  return (unsigned)__builtin_amdgcn_groupstaticsize();
#else
  return 0u;
#endif
}

__device__ __forceinline__ void tdm_load_1d(unsigned lds_byte_off, const void* gsrc, unsigned nelem) {
#if HAS_TDM
  unsigned long long ga = (unsigned long long)(uintptr_t)gsrc;
  v4u g0;
  g0[0] = 1u;                                    // count=1 (valid), user mode
  g0[1] = lds_byte_off;                          // lds_addr
  g0[2] = (unsigned)(ga & 0xFFFFFFFFu);          // global_addr[31:0]
  g0[3] = (unsigned)((ga >> 32) & 0x1FFFFFFu) | (2u << 30);  // global_addr[56:32], type=2
  v8i g1;
  g1[0] = 0x00010000;                            // data_size=1 (2B), mask=0
  g1[1] = (int)((nelem & 0xFFFFu) << 16);        // tensor_dim0[15:0] @ bits63:48
  g1[2] = (int)(((nelem >> 16) & 0xFFFFu) | (1u << 16)); // tensor_dim0 hi, tensor_dim1=1
  g1[3] = (int)((nelem & 0xFFFFu) << 16);        // tile_dim0 @ bits127:112
  g1[4] = 1;                                     // tile_dim1=1, tile_dim2=0
  g1[5] = (int)nelem;                            // tensor_dim0_stride lo32
  g1[6] = 0;
  g1[7] = 0;
  v4i gz = (v4i)0;
#if defined(__clang_major__) && (__clang_major__ >= 23)
  v8i gz8 = (v8i)0;
  __builtin_amdgcn_tensor_load_to_lds(g0, g1, gz, gz, gz8, 0);
#else
  __builtin_amdgcn_tensor_load_to_lds(g0, g1, gz, gz, 0);
#endif
#endif
}

__device__ __forceinline__ void tdm_wait() {
#if __has_builtin(__builtin_amdgcn_s_wait_tensorcnt)
  __builtin_amdgcn_s_wait_tensorcnt(0);
#else
  asm volatile("s_wait_tensorcnt 0x0" ::: "memory");
#endif
}

// fallback cooperative bf16 copy (used when !HAS_TDM)
__device__ __forceinline__ void coop_copy_bf16(bf16* dst, const bf16* src, int nelem,
                                               int tid, int nthr) {
  const uint4* s = (const uint4*)src;
  uint4* d = (uint4*)dst;
  for (int i = tid; i < nelem / 8; i += nthr) d[i] = s[i];
}

// -------------------- zero accumulators --------------------
__global__ void kz(float* __restrict__ p, int n) {
  int i = blockIdx.x * blockDim.x + threadIdx.x;
  if (i < n) p[i] = 0.f;
}

// -------------------- f32 -> bf16 convert --------------------
__global__ void kcvt(const float* __restrict__ src, bf16* __restrict__ dst, int n4) {
  int i = blockIdx.x * blockDim.x + threadIdx.x;
  if (i < n4) {
    float4 v = ((const float4*)src)[i];
    union { bf16 h[4]; uint2 u; } p;
    p.h[0] = (bf16)v.x; p.h[1] = (bf16)v.y; p.h[2] = (bf16)v.z; p.h[3] = (bf16)v.w;
    ((uint2*)dst)[i] = p.u;
  }
}

// ------------- precompute Wcomb = v_map_w @ mean_h(Wv) (bf16) and bcomb -------------
__global__ void k0_wcomb(const float* __restrict__ Wv, const float* __restrict__ bv,
                         const float* __restrict__ vmw, const float* __restrict__ vmb,
                         bf16* __restrict__ wcomb_bf, float* __restrict__ bcomb) {
  extern __shared__ float fsm[];
  float* Wvbar = fsm;  // 128x128 f32
  int tid = threadIdx.x;
  for (int i = tid; i < C * C; i += 256) {
    float s = 0.f;
#pragma unroll
    for (int h = 0; h < H; ++h) s += Wv[h * C * C + i];
    Wvbar[i] = s * (1.0f / H);
  }
  __syncthreads();
  for (int idx = tid; idx < C * C; idx += 256) {
    int o = idx >> 7, i = idx & 127;
    float s = 0.f;
    for (int d = 0; d < C; ++d) s += vmw[o * C + d] * Wvbar[d * C + i];
    wcomb_bf[idx] = (bf16)s;
  }
  if (tid < C) {
    float s = 0.f;
    for (int d = 0; d < C; ++d) {
      float bb = 0.f;
#pragma unroll
      for (int h = 0; h < H; ++h) bb += bv[h * C + d];
      s += vmw[tid * C + d] * (bb * (1.0f / H));
    }
    bcomb[tid] = s + vmb[tid];
  }
}

// ---- k1b: one-shot ktv^T (bf16) + kssum column tiles (16 cols, col0 = kssum) ----
__global__ void k1b_stage(const float* __restrict__ ktv, const float* __restrict__ kssum,
                          bf16* __restrict__ ktvT, bf16* __restrict__ kcol) {
  int i = blockIdx.x * blockDim.x + threadIdx.x;
  if (i < H * C * C) {
    int h = i >> 14, rem = i & 16383, d = rem >> 7, m = rem & 127;
    ktvT[i] = (bf16)ktv[(h << 14) + (m << 7) + d];   // ktvT[h][d][m] = ktv[h][m][d]
  }
  if (i < H * 16 * C) {
    int h = i >> 11, rem = i & 2047, j = rem >> 7, m = rem & 127;
    kcol[i] = (j == 0) ? (bf16)kssum[(h << 7) + m] : (bf16)0.f;
  }
}

// --------- phi feature map on register tile (C/D WMMA layout), returns scaled x^2 ---
__device__ __forceinline__ void phi_tile(v8f (&acc)[8], const float* __restrict__ bias,
                                         int lane, float inv_den, float* scale) {
  float sx2[8], sx4[8];
#pragma unroll
  for (int r = 0; r < 8; ++r) { sx2[r] = 0.f; sx4[r] = 0.f; }
#pragma unroll
  for (int st = 0; st < 8; ++st) {
    float b = bias[st * 16 + (lane & 15)];
#pragma unroll
    for (int r = 0; r < 8; ++r) {
      float x = acc[st][r] + b;
      x = (fmaxf(x, 0.f) + EPS_PHI) * inv_den;
      float x2 = x * x;
      acc[st][r] = x2;
      sx2[r] += x2;
      sx4[r] += x2 * x2;
    }
  }
#pragma unroll
  for (int r = 0; r < 8; ++r) {
#pragma unroll
    for (int m = 1; m < 16; m <<= 1) {
      sx2[r] += __shfl_xor(sx2[r], m);
      sx4[r] += __shfl_xor(sx4[r], m);
    }
    scale[r] = sqrtf(sx2[r]) / (sqrtf(sx4[r]) + EPS_FP);
  }
}

// batched 8-fragment B load + 8-WMMA burst (lets compiler overlap dscnt waits)
__device__ __forceinline__ void gemm_step(v16bf a, const bf16* B, int k0, int lane,
                                          v8f (&acc)[8]) {
  v16bf bf[8];
#pragma unroll
  for (int st = 0; st < 8; ++st) bf[st] = load_b_tr(B, k0, st * 16, lane);
#pragma unroll
  for (int st = 0; st < 8; ++st) acc[st] = wmma_bf16(a, bf[st], acc[st]);
}

// -------------------- k1: Q/K/V projection + phi + KtV/ks_sum ----------------------
__global__ void __launch_bounds__(256, 1)
k1_proj_phi_ktv(const float* __restrict__ q_in, const float* __restrict__ s_in,
                const bf16* __restrict__ wq_bf, const float* __restrict__ bq,
                const bf16* __restrict__ wk_bf, const float* __restrict__ bk,
                const bf16* __restrict__ wv_bf, const float* __restrict__ bv,
                const float* __restrict__ nscale,
                bf16* __restrict__ phiQ, float* __restrict__ ktv,
                float* __restrict__ kssum) {
  extern __shared__ char smem[];
  bf16* PKt = (bf16*)smem;         // phi_ks transposed [col][row]  32KB
  bf16* VTt = PKt + C * C;         // v transposed [col][row]       32KB
  bf16* WQs = VTt + C * C;         // Wq[h] [o][i]                  32KB
  bf16* WKs = WQs + C * C;
  bf16* WVs = WKs + C * C;

  const int tid = threadIdx.x;
  const int lane = tid & 31;
  const int wave = tid >> 5;
  const int half = lane >> 4;
  const int h = blockIdx.x / SPLITS;
  const int split = blockIdx.x % SPLITS;
  const int row0 = split * ROWS_PER_BLOCK;
  const int m0 = wave * 16;
  const float inv_den = 1.0f / (fabsf(nscale[0]) + EPS_PHI);

  // stage per-head weights via Tensor Data Mover (bf16, 32KB each)
#if HAS_TDM
  if (wave == 0) {
    unsigned base = lds_base_off();
    tdm_load_1d(base + (unsigned)((char*)WQs - smem), wq_bf + (size_t)h * C * C, C * C);
    tdm_load_1d(base + (unsigned)((char*)WKs - smem), wk_bf + (size_t)h * C * C, C * C);
    tdm_load_1d(base + (unsigned)((char*)WVs - smem), wv_bf + (size_t)h * C * C, C * C);
    tdm_wait();
  }
#else
  coop_copy_bf16(WQs, wq_bf + (size_t)h * C * C, C * C, tid, 256);
  coop_copy_bf16(WKs, wk_bf + (size_t)h * C * C, C * C, tid, 256);
  coop_copy_bf16(WVs, wv_bf + (size_t)h * C * C, C * C, tid, 256);
#endif
  __syncthreads();

  v8f kacc[8];
  float ksum[8];
#pragma unroll
  for (int st = 0; st < 8; ++st) {
    ksum[st] = 0.f;
#pragma unroll
    for (int r = 0; r < 8; ++r) kacc[st][r] = 0.f;
  }

  for (int t = 0; t < TILES; ++t) {
    const int rbase = row0 + t * TM;
    v8f acc[8];
    float scale[8];

    // ---------- qs = Xq @ Wq^T : A direct from global, B from LDS ----------
    {
      v16bf aq[4];
#pragma unroll
      for (int kk = 0; kk < 4; ++kk)
        aq[kk] = load_a_gf32(q_in + (size_t)rbase * C, m0, kk * 32, lane);
#pragma unroll
      for (int st = 0; st < 8; ++st)
#pragma unroll
        for (int r = 0; r < 8; ++r) acc[st][r] = 0.f;
#pragma unroll
      for (int kk = 0; kk < 4; ++kk) gemm_step(aq[kk], WQs, kk * 32, lane, acc);
    }
    phi_tile(acc, bq + h * C, lane, inv_den, scale);
#pragma unroll
    for (int st = 0; st < 8; ++st) {
      int col = st * 16 + (lane & 15);
#pragma unroll
      for (int r = 0; r < 8; ++r) {
        int row = rbase + m0 + r + 8 * half;
        phiQ[((size_t)h * NROWS + row) * C + col] = (bf16)(acc[st][r] * scale[r]);
      }
    }

    // ---------- shared A fragments for K/V from source_input ----------
    v16bf as[4];
#pragma unroll
    for (int kk = 0; kk < 4; ++kk)
      as[kk] = load_a_gf32(s_in + (size_t)rbase * C, m0, kk * 32, lane);

    // ---------- ks = Xs @ Wk^T, phi -> PKt (transposed) + ksum ----------
#pragma unroll
    for (int st = 0; st < 8; ++st)
#pragma unroll
      for (int r = 0; r < 8; ++r) acc[st][r] = 0.f;
#pragma unroll
    for (int kk = 0; kk < 4; ++kk) gemm_step(as[kk], WKs, kk * 32, lane, acc);
    phi_tile(acc, bk + h * C, lane, inv_den, scale);
#pragma unroll
    for (int st = 0; st < 8; ++st) {
      int col = st * 16 + (lane & 15);
      float tmp[8];
#pragma unroll
      for (int r = 0; r < 8; ++r) {
        tmp[r] = acc[st][r] * scale[r];
        ksum[st] += tmp[r];
      }
      store8_bf16(&PKt[col * C + m0 + 8 * half], tmp);
    }

    // ---------- v = Xs @ Wv^T + b -> VTt (transposed) ----------
#pragma unroll
    for (int st = 0; st < 8; ++st)
#pragma unroll
      for (int r = 0; r < 8; ++r) acc[st][r] = 0.f;
#pragma unroll
    for (int kk = 0; kk < 4; ++kk) gemm_step(as[kk], WVs, kk * 32, lane, acc);
#pragma unroll
    for (int st = 0; st < 8; ++st) {
      int col = st * 16 + (lane & 15);
      float b = bv[h * C + col];
      float tmp[8];
#pragma unroll
      for (int r = 0; r < 8; ++r) tmp[r] = acc[st][r] + b;
      store8_bf16(&VTt[col * C + m0 + 8 * half], tmp);
    }
    __syncthreads();  // PKt/VTt tiles ready

    // ---------- KtV += phi_ks^T @ v : A from PKt, B from VTt ----------
#pragma unroll
    for (int kk = 0; kk < 4; ++kk) {
      v16bf a = load_a_lds(PKt, m0, kk * 32, lane);
      gemm_step(a, VTt, kk * 32, lane, kacc);
    }
    __syncthreads();  // KtV reads done before next tile overwrites
  }

  // flush KtV partials
#pragma unroll
  for (int st = 0; st < 8; ++st) {
    int d = st * 16 + (lane & 15);
#pragma unroll
    for (int r = 0; r < 8; ++r) {
      int m = m0 + r + 8 * half;
      atomicAdd(&ktv[((size_t)h * C + m) * C + d], kacc[st][r]);
    }
  }
  // flush ks_sum partials
#pragma unroll
  for (int st = 0; st < 8; ++st) {
    float s = ksum[st] + __shfl_xor(ksum[st], 16);
    if (half == 0) atomicAdd(&kssum[h * C + st * 16 + lane], s);
  }
}

// -------------------- k2: numerator/denominator + vss + head mean + time -----------
__global__ void __launch_bounds__(256, 1)
k2_attn_out(const float* __restrict__ s_in, const bf16* __restrict__ phiQ,
            const bf16* __restrict__ ktvT, const bf16* __restrict__ kcol,
            const bf16* __restrict__ wcomb_bf, const float* __restrict__ bcomb,
            float* __restrict__ out) {
  extern __shared__ char smem[];
  bf16* Bb = (bf16*)smem;           // 128x128: wcomb [o][i], then ktv^T [d][m]
  bf16* Bs = Bb + C * C;            // 16x128: col0 = kssum[h], cols 1..15 = 0

  const int tid = threadIdx.x;
  const int lane = tid & 31;
  const int wave = tid >> 5;
  const int half = lane >> 4;
  const int m0 = wave * 16;
  const int rbase = blockIdx.x * TM;

  // ---- vss-mean GEMM: acc = Xs @ Wcomb^T + bcomb ----
#if HAS_TDM
  if (wave == 0) {
    tdm_load_1d(lds_base_off() + (unsigned)((char*)Bb - smem), wcomb_bf, C * C);
    tdm_wait();
  }
#else
  coop_copy_bf16(Bb, wcomb_bf, C * C, tid, 256);
#endif
  __syncthreads();
  v8f acc[8];
#pragma unroll
  for (int st = 0; st < 8; ++st)
#pragma unroll
    for (int r = 0; r < 8; ++r) acc[st][r] = 0.f;
#pragma unroll
  for (int kk = 0; kk < 4; ++kk) {
    v16bf a = load_a_gf32(s_in + (size_t)rbase * C, m0, kk * 32, lane);
    gemm_step(a, Bb, kk * 32, lane, acc);
  }
#pragma unroll
  for (int st = 0; st < 8; ++st) {
    float bc = bcomb[st * 16 + (lane & 15)];
#pragma unroll
    for (int r = 0; r < 8; ++r) acc[st][r] += bc;
  }

  for (int h = 0; h < H; ++h) {
    __syncthreads();  // prior GEMM reads of Bb/Bs done
    // stage ktv^T[h] and kssum column tile via TDM
#if HAS_TDM
    if (wave == 0) {
      unsigned base = lds_base_off();
      tdm_load_1d(base + (unsigned)((char*)Bb - smem), ktvT + (size_t)h * C * C, C * C);
      tdm_load_1d(base + (unsigned)((char*)Bs - smem), kcol + (size_t)h * 16 * C, 16 * C);
      tdm_wait();
    }
#else
    coop_copy_bf16(Bb, ktvT + (size_t)h * C * C, C * C, tid, 256);
    coop_copy_bf16(Bs, kcol + (size_t)h * 16 * C, 16 * C, tid, 256);
#endif
    __syncthreads();
    // numerator GEMM (B = ktv^T) + denominator column via WMMA
    v8f nacc[8], nden;
#pragma unroll
    for (int st = 0; st < 8; ++st)
#pragma unroll
      for (int r = 0; r < 8; ++r) nacc[st][r] = 0.f;
#pragma unroll
    for (int r = 0; r < 8; ++r) nden[r] = 0.f;
#pragma unroll
    for (int kk = 0; kk < 4; ++kk) {
      v16bf a = load_a_gbf(phiQ + ((size_t)h * NROWS + rbase) * C, m0, kk * 32, lane);
      gemm_step(a, Bb, kk * 32, lane, nacc);
      nden = wmma_bf16(a, load_b_tr(Bs, kk * 32, 0, lane), nden);
    }
#pragma unroll
    for (int r = 0; r < 8; ++r) {
      float den = __shfl(nden[r], lane & 16);  // col 0 of this lane's row group
      float dinv = (1.0f / H) / (den + EPS_PHI);
#pragma unroll
      for (int st = 0; st < 8; ++st) acc[st][r] += nacc[st][r] * dinv;
    }
  }

  // epilogue: row sumsq -> time coordinate, write [time, space] stride 129
  float ss[8];
#pragma unroll
  for (int r = 0; r < 8; ++r) ss[r] = 0.f;
#pragma unroll
  for (int st = 0; st < 8; ++st)
#pragma unroll
    for (int r = 0; r < 8; ++r) ss[r] += acc[st][r] * acc[st][r];
#pragma unroll
  for (int r = 0; r < 8; ++r) {
#pragma unroll
    for (int m = 1; m < 16; m <<= 1) ss[r] += __shfl_xor(ss[r], m);
  }
#pragma unroll
  for (int r = 0; r < 8; ++r) {
    int row = rbase + m0 + r + 8 * half;
    float* orow = out + (size_t)row * (C + 1);
    if ((lane & 15) == 0) orow[0] = sqrtf(ss[r] + 1.0f);  // MANIFOLD_K = 1
#pragma unroll
    for (int st = 0; st < 8; ++st) orow[1 + st * 16 + (lane & 15)] = acc[st][r];
  }
}

extern "C" void kernel_launch(void* const* d_in, const int* in_sizes, int n_in,
                              void* d_out, int out_size, void* d_ws, size_t ws_size,
                              hipStream_t stream) {
  const float* q_in = (const float*)d_in[0];
  const float* s_in = (const float*)d_in[1];
  const float* Wq = (const float*)d_in[2];
  const float* bq = (const float*)d_in[3];
  const float* Wk = (const float*)d_in[4];
  const float* bk = (const float*)d_in[5];
  const float* Wv = (const float*)d_in[6];
  const float* bv = (const float*)d_in[7];
  const float* vmw = (const float*)d_in[8];
  const float* vmb = (const float*)d_in[9];
  const float* nscale = (const float*)d_in[10];
  float* out = (float*)d_out;

  char* ws = (char*)d_ws;
  bf16* phiQ = (bf16*)ws;
  size_t off = (size_t)H * NROWS * C * sizeof(bf16);            // 67,108,864
  bf16* wq_bf = (bf16*)(ws + off);  off += (size_t)H * C * C * sizeof(bf16);
  bf16* wk_bf = (bf16*)(ws + off);  off += (size_t)H * C * C * sizeof(bf16);
  bf16* wv_bf = (bf16*)(ws + off);  off += (size_t)H * C * C * sizeof(bf16);
  bf16* ktvT_bf = (bf16*)(ws + off); off += (size_t)H * C * C * sizeof(bf16);
  bf16* kcol_bf = (bf16*)(ws + off); off += (size_t)H * 16 * C * sizeof(bf16);
  bf16* wcomb_bf = (bf16*)(ws + off); off += (size_t)C * C * sizeof(bf16);
  float* bcomb = (float*)(ws + off);  off += 128 * sizeof(float);
  float* ktv = (float*)(ws + off);    off += (size_t)H * C * C * sizeof(float);
  float* kssum = (float*)(ws + off);

  const int nz = H * C * C + H * C;  // ktv + kssum contiguous
  kz<<<(nz + 255) / 256, 256, 0, stream>>>(ktv, nz);
  const int nw4 = H * C * C / 4;
  kcvt<<<(nw4 + 255) / 256, 256, 0, stream>>>(Wq, wq_bf, nw4);
  kcvt<<<(nw4 + 255) / 256, 256, 0, stream>>>(Wk, wk_bf, nw4);
  kcvt<<<(nw4 + 255) / 256, 256, 0, stream>>>(Wv, wv_bf, nw4);
  k0_wcomb<<<1, 256, C * C * sizeof(float), stream>>>(Wv, bv, vmw, vmb, wcomb_bf, bcomb);
  k1_proj_phi_ktv<<<H * SPLITS, 256, 5 * C * C * sizeof(bf16), stream>>>(
      q_in, s_in, wq_bf, bq, wk_bf, bk, wv_bf, bv, nscale, phiQ, ktv, kssum);
  k1b_stage<<<(H * C * C + 255) / 256, 256, 0, stream>>>(ktv, kssum, ktvT_bf, kcol_bf);
  k2_attn_out<<<NROWS / TM, 256, (C * C + 16 * C) * sizeof(bf16), stream>>>(
      s_in, phiQ, ktvT_bf, kcol_bf, wcomb_bf, bcomb, out);
}
